// GeneratorDecoder_88098369176218
// MI455X (gfx1250) — matrix operand order repaired and checked by reading, not compiled, not run on hardware
//
#include <hip/hip_runtime.h>

// Problem constants (match reference setup_inputs()).
#define BB  32768
#define TT  128
#define HH  32
#define PP  7
#define NDD 8

typedef __attribute__((ext_vector_type(16))) _Float16 v16h;
typedef __attribute__((ext_vector_type(8)))  float    v8f;

__device__ __forceinline__ v8f wmma16(v16h a, v16h b, v8f c) {
  // D = A(16x32 f16) * B(32x16 f16) + C(16x16 f32)
  return __builtin_amdgcn_wmma_f32_16x16x32_f16(false, a, false, b, (short)0, c,
                                                false, false);
}

__device__ __forceinline__ v8f splat8(float x) {
  v8f r;
#pragma unroll
  for (int i = 0; i < 8; ++i) r[i] = x;
  return r;
}

// Fast sigmoid: v_exp + v_add + v_rcp (no IEEE-div fixup chain).
__device__ __forceinline__ float sigmoid_fast(float x) {
  return __builtin_amdgcn_rcpf(1.0f + __expf(-x));
}

// Native V_TANH_F32 on gfx1250 when the builtin exists; cheap fallback else.
#if __has_builtin(__builtin_amdgcn_tanhf)
__device__ __forceinline__ float tanh_fast(float x) {
  return __builtin_amdgcn_tanhf(x);
}
#else
__device__ __forceinline__ float tanh_fast(float x) {
  return 2.0f * __builtin_amdgcn_rcpf(1.0f + __expf(-2.0f * x)) - 1.0f;
}
#endif

__global__ __launch_bounds__(32)
void gen_decoder(const float* __restrict__ h_n,    // (1,B,H)
                 const float* __restrict__ c_n,    // (1,B,H)
                 const float* __restrict__ noise,  // (B,ND)
                 const float* __restrict__ gumbel, // (T,B,P)
                 const float* __restrict__ W_init, // (H, H+ND)
                 const float* __restrict__ b_init, // (H)
                 const float* __restrict__ W_ih,   // (4H, P)
                 const float* __restrict__ W_hh,   // (4H, H)
                 const float* __restrict__ b_ih,   // (4H)
                 const float* __restrict__ b_hh,   // (4H)
                 const float* __restrict__ W_head, // (P, H)
                 const float* __restrict__ b_head, // (P)
                 float* __restrict__ out_logits,   // (B,T,P)
                 float* __restrict__ out_samples)  // (B,T,P)
{
  // [0] = W_ih B-tiles (K rows 0..6 = weights, row 7 = gate bias, 8..31 = 0),
  // [1] = W_hh B-tiles. 16 KB, read-only after init.
  __shared__ v16h lds_Bw[2][8][32];
  __shared__ alignas(16) _Float16 lds_h[16 * HH];  // h tile, row-major f16

  const int lane = threadIdx.x & 31;
  const int l16  = lane & 15;   // N / class / A-row index
  const int hB   = lane >> 4;   // lane half
  const int Kb   = hB ? 16 : 0; // K base for B layout
  const int Ka   = hB ? 8 : 0;  // K base for A layout half-split
  const int rowBase = blockIdx.x * 16;

  // ---- stage W_ih / W_hh B-tiles into LDS (clamped loads, no branches) ----
#pragma unroll
  for (int n = 0; n < 8; ++n) {
    const int r = 16 * n + l16;                 // gate column = weight row
    const float bsum = b_ih[r] + b_hh[r];       // bias folded into K=7 row
    v16h bi, bh;
#pragma unroll
    for (int e = 0; e < 16; ++e) {
      const int K  = Kb + e;
      const int Kc = (K < PP) ? K : 0;
      const float wi = W_ih[r * PP + Kc];       // always-valid address
      bi[e] = (K < PP) ? (_Float16)wi
                       : ((K == PP) ? (_Float16)bsum : (_Float16)0.0f);
      bh[e] = (_Float16)W_hh[r * HH + K];
    }
    lds_Bw[0][n][lane] = bi;
    lds_Bw[1][n][lane] = bh;
  }

  // ---- head as A-matrix: row M = class (rows 7..15 zero), K = hidden unit ----
  const int nr = (l16 < PP) ? l16 : 0;
  v16h Awh;
#pragma unroll
  for (int e = 0; e < 8; ++e) {
    const float w = W_head[nr * HH + Kb + e];
    Awh[e] = (l16 < PP) ? (_Float16)w : (_Float16)0.0f;
  }
#pragma unroll
  for (int e = 0; e < 8; ++e) {
    const float w = W_head[nr * HH + Kb + 8 + e];
    Awh[8 + e] = (l16 < PP) ? (_Float16)w : (_Float16)0.0f;
  }
  // head bias as per-VGPR C vector: D[class j, row] += b_head[j]
  v8f bc8;
#pragma unroll
  for (int j = 0; j < 8; ++j) bc8[j] = (j < PP) ? b_head[j] : 0.0f;

  // ---- h0 = [h_n ; noise] @ W_init^T + b_init  (2 K-chunks x 2 N-tiles) ----
  v16h A1, A2;
  {
    const int r = rowBase + l16;
#pragma unroll
    for (int e = 0; e < 8; ++e)
      A1[e] = (_Float16)h_n[r * HH + Kb + e];
#pragma unroll
    for (int e = 0; e < 8; ++e)
      A1[8 + e] = (_Float16)h_n[r * HH + Kb + 8 + e];
#pragma unroll
    for (int e = 0; e < 16; ++e) A2[e] = (_Float16)0.0f;
#pragma unroll
    for (int e = 0; e < NDD; ++e) {
      const float w = noise[r * NDD + e];       // unconditional load
      A2[e] = hB ? (_Float16)0.0f : (_Float16)w;
    }
  }

  float hf[2][8]; // h state, C/D layout (g: units 0-15 / 16-31)
  float cs[2][8]; // cell state
#pragma unroll
  for (int g = 0; g < 2; ++g) {
    const int u = 16 * g + l16; // hidden unit = W_init row
    v16h B1, B2;
#pragma unroll
    for (int e = 0; e < 16; ++e)
      B1[e] = (_Float16)W_init[u * (HH + NDD) + Kb + e];
#pragma unroll
    for (int e = 0; e < 16; ++e) {
      const int K  = Kb + e;
      const int Kc = (K < NDD) ? K : 0;
      const float w = W_init[u * (HH + NDD) + HH + Kc];
      B2[e] = (K < NDD) ? (_Float16)w : (_Float16)0.0f;
    }
    v8f acc = wmma16(A2, B2, wmma16(A1, B1, splat8(b_init[u])));
#pragma unroll
    for (int j = 0; j < 8; ++j) hf[g][j] = acc[j];
#pragma unroll
    for (int j = 0; j < 8; ++j)
      cs[g][j] = c_n[(size_t)(rowBase + j + hB * 8) * HH + u];
  }

  // y0 == 0, but bias row K=7 must be hot from step 0.
  v16h Ay;
#pragma unroll
  for (int e = 0; e < 16; ++e) Ay[e] = (_Float16)0.0f;
  if (!hB) Ay[PP] = (_Float16)1.0f;

  // shared zero C operand for the gate WMMA chains (loop-invariant)
  const v8f zero8 = splat8(0.0f);

  // push h0 through LDS to get A-layout f16 tile
  __syncthreads();
#pragma unroll
  for (int g = 0; g < 2; ++g)
#pragma unroll
    for (int j = 0; j < 8; ++j)
      lds_h[(j + hB * 8) * HH + 16 * g + l16] = (_Float16)hf[g][j];
  __syncthreads();
  v16h Ah;
  {
    const _Float16* p = &lds_h[l16 * HH + Kb];
    union { uint4 q[2]; v16h v; } cv;
    cv.q[0] = *(const uint4*)p;        // K Kb..Kb+7
    cv.q[1] = *(const uint4*)(p + 8);  // K Kb+8..Kb+15
    Ah = cv.v;
  }

  const float EPSg = 1e-20f;

#pragma unroll 1
  for (int t = 0; t < TT; ++t) {
    // gates = [y;1]@[W_ih;bias]^T + h@W_hh^T : 16 chained WMMAs, B from LDS
    v8f acc[8];
#pragma unroll
    for (int n = 0; n < 8; ++n) {
      const v16h bih = lds_Bw[0][n][lane];
      const v16h bhh = lds_Bw[1][n][lane];
      acc[n] = wmma16(Ah, bhh, wmma16(Ay, bih, zero8));
    }

    // LSTM cell: tiles {0,1}=i {2,3}=f {4,5}=g {6,7}=o share coordinates
#pragma unroll
    for (int g = 0; g < 2; ++g)
#pragma unroll
      for (int j = 0; j < 8; ++j) {
        const float iv = sigmoid_fast(acc[0 + g][j]);
        const float fv = sigmoid_fast(acc[2 + g][j]);
        const float gv = tanh_fast(acc[4 + g][j]);
        const float ov = sigmoid_fast(acc[6 + g][j]);
        const float cc = fv * cs[g][j] + iv * gv;
        cs[g][j] = cc;
        hf[g][j] = ov * tanh_fast(cc);
      }

    // h (C/D f32) -> LDS -> {A-layout tile, B-layout (h^T) tile}
    __syncthreads(); // WAR: prior lds_h readers done
#pragma unroll
    for (int g = 0; g < 2; ++g)
#pragma unroll
      for (int j = 0; j < 8; ++j)
        lds_h[(j + hB * 8) * HH + 16 * g + l16] = (_Float16)hf[g][j];
    __syncthreads();
    v16h Bh;
    {
      const _Float16* p = &lds_h[l16 * HH + Kb];
      union { uint4 q[2]; v16h v; } cv;
      cv.q[0] = *(const uint4*)p;
      cv.q[1] = *(const uint4*)(p + 8);
      Ah = cv.v;
      const _Float16* pb = &lds_h[l16 * HH + Kb]; // B[K,N]=h[row N][unit K]
      union { uint4 q[2]; v16h v; } cb;
      cb.q[0] = *(const uint4*)pb;       // K Kb..Kb+7
      cb.q[1] = *(const uint4*)(pb + 8); // K Kb+8..Kb+15
      Bh = cb.v;
    }

    // transposed head: D[class, row] = W_head x h^T + b_head
    // => lane L<16 holds all logits of batch row L in lt[0..6]
    const v8f lt = wmma16(Awh, Bh, bc8);

    // row-per-lane Gumbel hard-max sampling (lanes 0..15 own rows 0..15)
    int k = 0;
    if (lane < 16) {
      const int row = rowBase + lane;
      const float* gu = gumbel + ((size_t)t * BB + row) * PP; // 7 contiguous
      float z[PP];
#pragma unroll
      for (int c = 0; c < PP; ++c) {
        const float gn = -__logf(-__logf(gu[c] + EPSg) + EPSg);
        z[c] = lt[c] + gn;
      }
      float m = z[0];
#pragma unroll
      for (int c = 1; c < PP; ++c)
        if (z[c] > m) { m = z[c]; k = c; } // strict > == first-occurrence argmax
      float p[PP], s = 0.0f;
#pragma unroll
      for (int c = 0; c < PP; ++c) { p[c] = __expf(z[c] - m); s += p[c]; }
      const float rs = __builtin_amdgcn_rcpf(s);
      float* ol = out_logits  + ((size_t)row * TT + t) * PP; // 7 contiguous
      float* os = out_samples + ((size_t)row * TT + t) * PP;
#pragma unroll
      for (int c = 0; c < PP; ++c) {
        const float ys = p[c] * rs;
        ol[c] = lt[c];
        os[c] = ((c == k ? 1.0f : 0.0f) - ys) + ys; // straight-through value
      }
      if (t + 1 < TT) __builtin_prefetch(gu + (size_t)BB * PP, 0, 0);
    }

    // broadcast row l16's argmax to both lane halves (no LDS, no barrier)
    const int kb = __shfl(k, l16, 32);

    // one-hot (+ bias row) A tile for next step's [y;1] @ [W_ih;bias]^T
#pragma unroll
    for (int e = 0; e < 16; ++e) Ay[e] = (_Float16)0.0f;
    if (!hB) {
#pragma unroll
      for (int e = 0; e < PP; ++e)
        Ay[e] = (e == kb) ? (_Float16)1.0f : (_Float16)0.0f;
      Ay[PP] = (_Float16)1.0f; // constant bias row
    }
  }
}

extern "C" void kernel_launch(void* const* d_in, const int* in_sizes, int n_in,
                              void* d_out, int out_size, void* d_ws, size_t ws_size,
                              hipStream_t stream) {
  (void)in_sizes; (void)n_in; (void)out_size; (void)d_ws; (void)ws_size;
  const float* h_n   = (const float*)d_in[0];
  const float* c_n   = (const float*)d_in[1];
  const float* noise = (const float*)d_in[2];
  const float* gum   = (const float*)d_in[3];
  const float* Wi    = (const float*)d_in[4];
  const float* bi    = (const float*)d_in[5];
  const float* Wih   = (const float*)d_in[6];
  const float* Whh   = (const float*)d_in[7];
  const float* bih   = (const float*)d_in[8];
  const float* bhh   = (const float*)d_in[9];
  const float* Whd   = (const float*)d_in[10];
  const float* bhd   = (const float*)d_in[11];
  float* out_logits  = (float*)d_out;
  float* out_samples = out_logits + (size_t)BB * TT * PP;

  dim3 grid(BB / 16), block(32); // one wave32 per 16-row tile, persistent over T
  gen_decoder<<<grid, block, 0, stream>>>(h_n, c_n, noise, gum, Wi, bi, Wih,
                                          Whh, bih, bhh, Whd, bhd,
                                          out_logits, out_samples);
}
